// Model_74277164417213
// MI455X (gfx1250) — matrix operand order repaired and checked by reading, not compiled, not run on hardware
//
#include <hip/hip_runtime.h>
#include <math.h>

#define B_SZ 1024
#define T_SZ 512
#define K_SZ 64

#define LOG2E 1.44269504088896340736f
#define LN2   0.69314718055994530942f

typedef __attribute__((ext_vector_type(16))) _Float16 v16h;
typedef __attribute__((ext_vector_type(8)))  _Float16 v8h;
typedef __attribute__((ext_vector_type(8)))  float    v8f;
typedef __attribute__((ext_vector_type(4)))  int      v4i;

#define AS1 __attribute__((address_space(1)))
#define AS3 __attribute__((address_space(3)))

#if __has_builtin(__builtin_amdgcn_global_load_async_to_lds_b128) && \
    __has_builtin(__builtin_amdgcn_s_wait_asynccnt)
#define USE_ASYNC 1
#else
#define USE_ASYNC 0
#endif

__device__ __forceinline__ float fast_log2(float x) {
#if __has_builtin(__builtin_amdgcn_logf)
    return __builtin_amdgcn_logf(x);      // raw v_log_f32 (log2)
#else
    return log2f(x);                       // host-pass-safe fallback
#endif
}
__device__ __forceinline__ float fast_exp2(float x) {
#if __has_builtin(__builtin_amdgcn_exp2f)
    return __builtin_amdgcn_exp2f(x);     // raw v_exp_f32 (exp2)
#else
    return exp2f(x);                       // host-pass-safe fallback
#endif
}

// ---------------------------------------------------------------------------
// Kernel 1: P16[j][k] = (f16) exp(A[j][k]) = 2^(A*log2e)   (64x64, once)
// ---------------------------------------------------------------------------
__global__ void expA_kernel(const float* __restrict__ A, _Float16* __restrict__ P16) {
    int i = blockIdx.x * blockDim.x + threadIdx.x;
    if (i < K_SZ * K_SZ) P16[i] = (_Float16)__expf(A[i]);
}

// ---------------------------------------------------------------------------
// Kernel 2: CRF forward recursion in base-2 domain. One wave owns 16 rows.
// beta kept in registers in WMMA C/D layout:
//   af[ct][v] = beta[row = v + 8*(lane>>4)][col = ct*16 + (lane&15)]
// Per step: rowmax -> u=exp2(beta-m) (f16, LDS) -> 8x v_wmma_f32_16x16x32_f16
//           -> beta' = fma(logit, log2e, m + log2(s))
// Emissions for step t+1 are double-buffered into LDS via async-to-LDS DMA.
// ---------------------------------------------------------------------------
__launch_bounds__(32)
__global__ void crf_forward_kernel(const float* __restrict__ logits,
                                   const _Float16* __restrict__ P16,
                                   float* __restrict__ logZ) {
    const int lane    = threadIdx.x;   // 0..31
    const int half    = lane >> 4;     // 0 or 1
    const int lm      = lane & 15;
    const int rowbase = blockIdx.x * 16;
    const size_t rstride = (size_t)T_SZ * K_SZ;

    __shared__ __align__(16) _Float16 uS[16 * K_SZ];       // u staging, 2 KB
#if USE_ASYNC
    __shared__ __align__(16) float lbuf[2][16 * K_SZ];     // logit double buffer, 8 KB
#endif

    // --- Load B fragments of P = exp(A). B layout (32x16 f16):
    //     element e of lane L holds P[kc*32 + (L<16?0:16) + e][ct*16 + (L&15)]
    v16h bfrag[8];
    #pragma unroll
    for (int ct = 0; ct < 4; ++ct) {
        #pragma unroll
        for (int kc = 0; kc < 2; ++kc) {
            v16h f;
            const int col = ct * 16 + lm;
            const int jb  = kc * 32 + half * 16;
            #pragma unroll
            for (int e = 0; e < 16; ++e) f[e] = P16[(jb + e) * K_SZ + col];
            bfrag[ct * 2 + kc] = f;
        }
    }

    // --- beta(t=0) = logits[:,0,:] * log2e in D layout
    v8f af[4];
    #pragma unroll
    for (int v = 0; v < 8; ++v) {
        const float* bp = logits + (size_t)(rowbase + v + 8 * half) * rstride + lm;
        #pragma unroll
        for (int ct = 0; ct < 4; ++ct) af[ct][v] = bp[ct * 16] * LOG2E;
    }

#if USE_ASYNC
    // Stage logits[rows, t, :] (16 rows x 64 f32 = 4 KB) into lbuf[t&1]:
    // instruction i: lane covers row r = i*2 + half, floats [lm*4 .. lm*4+3].
    auto issue_step = [&](int t) {
        const float* g0 = logits + (size_t)(rowbase + half) * rstride
                                 + (size_t)t * K_SZ + lm * 4;
        char* l0 = (char*)&lbuf[t & 1][0] + (size_t)(half * 16 + lm) * 16;
        #pragma unroll
        for (int i = 0; i < 8; ++i) {
            __builtin_amdgcn_global_load_async_to_lds_b128(
                (AS1 v4i*)(g0 + (size_t)(i * 2) * rstride),
                (AS3 v4i*)(l0 + i * 512), 0, 0);
        }
    };
    issue_step(1);
#endif

    // --- forward recursion
    for (int t = 1; t < T_SZ; ++t) {
#if USE_ASYNC
        __builtin_amdgcn_s_wait_asynccnt(0);   // lbuf[t&1] ready
        if (t + 1 < T_SZ) issue_step(t + 1);   // overlap next tile with compute
#else
        if (t + 1 < T_SZ) {
            const float* pf = logits + (size_t)(rowbase + lm) * rstride
                                     + (size_t)(t + 1) * K_SZ + half * 32;
            __builtin_prefetch(pf, 0, 1);
        }
#endif

        // per-row max over the 64 states (4 tiles local + butterfly over 16 lanes)
        float rmax[8];
        #pragma unroll
        for (int v = 0; v < 8; ++v) {
            float mv = fmaxf(fmaxf(af[0][v], af[1][v]), fmaxf(af[2][v], af[3][v]));
            #pragma unroll
            for (int off = 1; off < 16; off <<= 1)
                mv = fmaxf(mv, __shfl_xor(mv, off, 32));
            rmax[v] = mv;
        }

        // u = 2^(beta - m) -> LDS, row-major f16 [16][64]
        #pragma unroll
        for (int v = 0; v < 8; ++v) {
            const int row = v + 8 * half;
            #pragma unroll
            for (int ct = 0; ct < 4; ++ct)
                uS[row * K_SZ + ct * 16 + lm] = (_Float16)fast_exp2(af[ct][v] - rmax[v]);
        }
        __syncthreads();

        // A fragments from LDS. A layout (16x32 f16): lane L is row (L&15);
        // element e -> K = kc*32 + half*16 + e
        v16h afrag[2];
        #pragma unroll
        for (int kc = 0; kc < 2; ++kc) {
            const int base = lm * K_SZ + kc * 32 + half * 16;
            v8h lo = *(const v8h*)&uS[base];
            v8h hi = *(const v8h*)&uS[base + 8];
            v16h a;
            #pragma unroll
            for (int e = 0; e < 8; ++e) { a[e] = lo[e]; a[e + 8] = hi[e]; }
            afrag[kc] = a;
        }
        __syncthreads();

        // s = u @ P (4 column tiles, K split in two 32-chunks), then
        // beta' = fma(logit, log2e, m + log2(s))   (result already in D layout)
        #pragma unroll
        for (int ct = 0; ct < 4; ++ct) {
            v8f acc = {0.f, 0.f, 0.f, 0.f, 0.f, 0.f, 0.f, 0.f};
            acc = __builtin_amdgcn_wmma_f32_16x16x32_f16(
                false, afrag[0], false, bfrag[ct * 2 + 0], (short)0, acc, false, false);
            acc = __builtin_amdgcn_wmma_f32_16x16x32_f16(
                false, afrag[1], false, bfrag[ct * 2 + 1], (short)0, acc, false, false);
            #pragma unroll
            for (int v = 0; v < 8; ++v) {
#if USE_ASYNC
                const float lg = lbuf[t & 1][(v + 8 * half) * K_SZ + ct * 16 + lm];
#else
                const float lg = logits[(size_t)(rowbase + v + 8 * half) * rstride
                                        + (size_t)t * K_SZ + ct * 16 + lm];
#endif
                af[ct][v] = fmaf(lg, LOG2E, rmax[v] + fast_log2(acc[v]));
            }
        }
    }

    // --- log_Z[b] = ln2 * log2sumexp2_k beta_T[b,k]   (natural log out)
    #pragma unroll
    for (int v = 0; v < 8; ++v) {
        float mv = fmaxf(fmaxf(af[0][v], af[1][v]), fmaxf(af[2][v], af[3][v]));
        #pragma unroll
        for (int off = 1; off < 16; off <<= 1)
            mv = fmaxf(mv, __shfl_xor(mv, off, 32));
        float s = fast_exp2(af[0][v] - mv) + fast_exp2(af[1][v] - mv)
                + fast_exp2(af[2][v] - mv) + fast_exp2(af[3][v] - mv);
        #pragma unroll
        for (int off = 1; off < 16; off <<= 1)
            s += __shfl_xor(s, off, 32);
        if (lm == 0) logZ[rowbase + v + 8 * half] = LN2 * (mv + fast_log2(s));
    }
}

// ---------------------------------------------------------------------------
// Kernel 3: gold path score + combine.
// loss += (logZ[b] - sum_t logits[b,t,g_t] - sum_t A[g_{t-1},g_t]) / B
// ---------------------------------------------------------------------------
__global__ void gold_loss_kernel(const float* __restrict__ logits,
                                 const float* __restrict__ A,
                                 const int* __restrict__ gold,
                                 const float* __restrict__ logZ,
                                 float* __restrict__ out) {
    const int b   = blockIdx.x;
    const int tid = threadIdx.x;       // 256 threads
    __shared__ float red[256];

    float acc = 0.f;
    for (int t = tid; t < T_SZ; t += 256) {
        const int g = gold[b * T_SZ + t];
        acc += logits[(size_t)b * T_SZ * K_SZ + (size_t)t * K_SZ + g];
        if (t > 0) {
            const int gp = gold[b * T_SZ + t - 1];
            acc += A[gp * K_SZ + g];
        }
    }
    red[tid] = acc;
    __syncthreads();
    for (int s = 128; s > 0; s >>= 1) {
        if (tid < s) red[tid] += red[tid + s];
        __syncthreads();
    }
    if (tid == 0) atomicAdd(out, (logZ[b] - red[0]) * (1.0f / B_SZ));
}

// ---------------------------------------------------------------------------
extern "C" void kernel_launch(void* const* d_in, const int* in_sizes, int n_in,
                              void* d_out, int out_size, void* d_ws, size_t ws_size,
                              hipStream_t stream) {
    const float* logits = (const float*)d_in[0];   // [B,T,K] f32
    const float* A      = (const float*)d_in[1];   // [K,K]   f32
    const int*   gold   = (const int*)d_in[2];     // [B,T]   int

    float*     out  = (float*)d_out;                                   // scalar
    _Float16*  P16  = (_Float16*)d_ws;                                 // 8 KB
    float*     logZ = (float*)((char*)d_ws + K_SZ * K_SZ * sizeof(_Float16)); // 4 KB

    (void)hipMemsetAsync(d_out, 0, sizeof(float), stream);
    expA_kernel<<<(K_SZ * K_SZ + 255) / 256, 256, 0, stream>>>(A, P16);
    crf_forward_kernel<<<B_SZ / 16, 32, 0, stream>>>(logits, P16, logZ);
    gold_loss_kernel<<<B_SZ, 256, 0, stream>>>(logits, A, gold, logZ, out);
}
